// SingleGlobalHeadProbEncoder_43662637531770
// MI455X (gfx1250) — matrix operand order repaired and checked by reading, not compiled, not run on hardware
//
#include <hip/hip_runtime.h>
#include <hip/hip_bf16.h>

// SingleGlobalHeadProbEncoder — fused WMMA-bf16 implementation for gfx1250.
// B=8, L=512, D=64, HEADS=8, G=64, ITER=3. step=1, damp=0 => blend == overwrite.

#define NB 8
#define NL 512
#define ND 64
#define NC 8
#define NG 64

typedef __attribute__((ext_vector_type(16))) __bf16 v16bf;
typedef __attribute__((ext_vector_type(8)))  float  v8f;

__device__ inline __bf16 f2bf(float f) { return (__bf16)f; }  // native cvt on gfx1250

__device__ inline v8f zero8() {
    v8f v;
#pragma unroll
    for (int t = 0; t < 8; ++t) v[t] = 0.f;
    return v;
}

__device__ inline v8f wmma_bf(v16bf a, v16bf b, v8f c) {
    return __builtin_amdgcn_wmma_f32_16x16x32_bf16(false, a, false, b, (short)0, c,
                                                   false, false);
}

// A fragment (16x32, 16-bit) from row-major src: lane&15 = row M,
// halves 0..7 = K (k0 + kh .. +7), halves 8..15 = K (k0 + kh + 16 .. +23),
// kh = 8*(lane>=16).  Matches ISA 7.12.2 "16-bit A-Matrix 16x32" table.
__device__ inline v16bf load_aRM(const __bf16* a, int lda, int k0) {
    int lane = threadIdx.x & 31;
    const __bf16* p = a + (lane & 15) * lda + k0 + ((lane >> 4) << 3);
    v16bf f;
#pragma unroll
    for (int t = 0; t < 8; ++t) { f[t] = p[t]; f[t + 8] = p[t + 16]; }
    return f;
}

// B fragment (32x16, 16-bit) from a TRANSPOSED store bt[n][k] (row n = col n of B):
// lane&15 = column N, halves 0..7 = K (k0 + kh .. +7), 8..15 = K (k0 + kh + 16 .. +23), kh = 8*(lane>=16).
__device__ inline v16bf load_bT(const __bf16* bt, int ldb, int k0) {
    int lane = threadIdx.x & 31;
    const __bf16* p = bt + (lane & 15) * ldb + k0 + ((lane >> 4) << 3);
    v16bf f;
#pragma unroll
    for (int t = 0; t < 8; ++t) { f[t] = p[t]; f[t + 8] = p[t + 16]; }
    return f;
}

// Async-copy one 2KB tile (16 rows x 64 bf16) global -> LDS using the gfx1250
// ASYNCcnt path.  Each of the 4 instructions moves 32 lanes x 16B = 512B; the
// instruction offset applies to both the LDS and global addresses (ISA 08 §4.4).
__device__ inline void stage_tile_async(__bf16* lds_dst, const __bf16* gsrc) {
    int lane = threadIdx.x & 31;
    unsigned loff = (unsigned)(size_t)(void*)lds_dst + lane * 16;
    const char* g = (const char*)gsrc + lane * 16;
    asm volatile("global_load_async_to_lds_b128 %0, %1, off"
                 :: "v"(loff), "v"(g) : "memory");
    asm volatile("global_load_async_to_lds_b128 %0, %1, off offset:512"
                 :: "v"(loff), "v"(g) : "memory");
    asm volatile("global_load_async_to_lds_b128 %0, %1, off offset:1024"
                 :: "v"(loff), "v"(g) : "memory");
    asm volatile("global_load_async_to_lds_b128 %0, %1, off offset:1536"
                 :: "v"(loff), "v"(g) : "memory");
}

// ---------------- prep: pack ternary / global_w into bf16 transposed layouts ----
// W1T[k][c*64+e][a] = ternary[k,a,e,c]   (for t1 = q_z @ T)
// W2T[k][c*64+a][e] = ternary[k,a,e,c]   (for t2 = q_z @ T over e)
__global__ void k_prep(const float* __restrict__ tern, const float* __restrict__ gw,
                       __bf16* W1T, __bf16* W2T, __bf16* gwb, __bf16* gwT) {
    int idx = blockIdx.x * 256 + threadIdx.x;
    if (idx < 2 * 512 * 64) {
        int k = idx >> 15;
        int rem = idx & 32767;
        int col = rem >> 6;    // 0..511
        int last = rem & 63;
        int c = col >> 6, sub = col & 63;
        W1T[idx] = f2bf(tern[(((k * 64 + last) * 64 + sub) * 8) + c]); // a=last,e=sub
        W2T[idx] = f2bf(tern[(((k * 64 + sub) * 64 + last) * 8) + c]); // a=sub,e=last
    }
    if (idx < 64 * 64) {
        int g = idx >> 6, a = idx & 63;
        float v = gw[idx];            // global_w[g][a][0]
        gwb[g * 64 + a] = f2bf(v);
        gwT[a * 64 + g] = f2bf(v);
    }
}

__global__ void k_fzinit(float* out, const float* __restrict__ x) {
    int idx = blockIdx.x * 256 + threadIdx.x;
    if (idx < NB * NL * ND) out[idx] = x[idx];
}

// ---------------- q_z = softmax(Z) * m1, stored bf16 [B,L,64] ------------------
__global__ void k_qz(const float* __restrict__ zin, const int* __restrict__ mask,
                     __bf16* qzb) {
    int r = blockIdx.x, lane = threadIdx.x;
    float v0 = zin[r * 64 + lane];
    float v1 = zin[r * 64 + 32 + lane];
    float m = fmaxf(v0, v1);
#pragma unroll
    for (int off = 16; off; off >>= 1) m = fmaxf(m, __shfl_xor(m, off, 32));
    float e0 = __expf(v0 - m), e1 = __expf(v1 - m);
    float s = e0 + e1;
#pragma unroll
    for (int off = 16; off; off >>= 1) s += __shfl_xor(s, off, 32);
    float inv = ((mask[r] != 0) ? 1.f : 0.f) / s;
    qzb[r * 64 + lane] = f2bf(e0 * inv);
    qzb[r * 64 + 32 + lane] = f2bf(e1 * inv);
}

// ------- t1[k][b][i][c*64+e], t1T[k][b][c*64+e][i], t2T[k][b][c*64+a][j] -------
__global__ void k_t1t2(const __bf16* __restrict__ qzb, const __bf16* __restrict__ W1T,
                       const __bf16* __restrict__ W2T, __bf16* t1, __bf16* t1T,
                       __bf16* t2T) {
    int mt = blockIdx.x, nt = blockIdx.y, kb = blockIdx.z;
    int k = kb >> 3, b = kb & 7;
    int lane = threadIdx.x;
    const __bf16* A = qzb + ((size_t)(b * 512 + mt * 16)) * 64;
    v16bf a0 = load_aRM(A, 64, 0), a1 = load_aRM(A, 64, 32);
    const __bf16* B1 = W1T + ((size_t)(k * 512 + nt * 16)) * 64;
    const __bf16* B2 = W2T + ((size_t)(k * 512 + nt * 16)) * 64;
    v8f o1 = zero8(), o2 = zero8();
    o1 = wmma_bf(a0, load_bT(B1, 64, 0), o1);
    o1 = wmma_bf(a1, load_bT(B1, 64, 32), o1);
    o2 = wmma_bf(a0, load_bT(B2, 64, 0), o2);
    o2 = wmma_bf(a1, load_bT(B2, 64, 32), o2);
    int n = lane & 15, hb = (lane >> 4) << 3;
    size_t base = (size_t)(k * 8 + b) * 512;
#pragma unroll
    for (int r = 0; r < 8; ++r) {
        int i = mt * 16 + r + hb;
        int col = nt * 16 + n;
        __bf16 h1 = f2bf(o1[r]);
        t1[(base + i) * 512 + col] = h1;
        t1T[(base + col) * 512 + i] = h1;
        t2T[(base + col) * 512 + i] = f2bf(o2[r]);
    }
}

// ------- fused scores + softmax + F1 + masked-P^T spill, per (b, c, i-tile) ----
__global__ void k_attn(const __bf16* __restrict__ t1, const __bf16* __restrict__ qzb,
                       const __bf16* __restrict__ t2T, const int* __restrict__ mask,
                       __bf16* PTm, float* fz) {
    __shared__ float Sb[16 * 512];
    __shared__ float rmax[16], rinv[16];
    __shared__ __bf16 Qt[2][16 * 64];   // double-buffered q_z j-tiles (async staged)
    int it = blockIdx.x, c = blockIdx.y, b = blockIdx.z, lane = threadIdx.x;
    int n = lane & 15, hb = (lane >> 4) << 3;

    const __bf16* A0 = t1 + ((size_t)(0 * 8 + b) * 512 + it * 16) * 512 + c * 64;
    const __bf16* A1 = t1 + ((size_t)(1 * 8 + b) * 512 + it * 16) * 512 + c * 64;
    v16bf a00 = load_aRM(A0, 512, 0), a01 = load_aRM(A0, 512, 32);
    v16bf a10 = load_aRM(A1, 512, 0), a11 = load_aRM(A1, 512, 32);

    const __bf16* Qg = qzb + (size_t)b * 512 * 64;
    stage_tile_async(Qt[0], Qg);                       // prologue: tile jt=0

    // scores: j>i uses t1[k=0], j<i uses t1[k=1], diag = 0
    for (int jt = 0; jt < 32; ++jt) {
        if (jt < 31) {
            stage_tile_async(Qt[(jt + 1) & 1], Qg + (size_t)(jt + 1) * 16 * 64);
            asm volatile("s_wait_asynccnt 0x4" ::: "memory");   // tile jt landed
        } else {
            asm volatile("s_wait_asynccnt 0x0" ::: "memory");
        }
        const __bf16* Bq = Qt[jt & 1];
        v16bf b0 = load_bT(Bq, 64, 0), b1 = load_bT(Bq, 64, 32);
        v8f s0 = zero8(), s1 = zero8();
        if (jt >= it) { s0 = wmma_bf(a00, b0, s0); s0 = wmma_bf(a01, b1, s0); }
        if (jt <= it) { s1 = wmma_bf(a10, b0, s1); s1 = wmma_bf(a11, b1, s1); }
        int j = jt * 16 + n;
#pragma unroll
        for (int r = 0; r < 8; ++r) {
            int il = r + hb, i = it * 16 + il;
            Sb[il * 512 + j] = (j > i) ? s0[r] : ((j < i) ? s1[r] : 0.f);
        }
    }
    __syncthreads();
    if (lane < 16) {
        float m = -1e30f;
        for (int j = 0; j < 512; ++j) m = fmaxf(m, Sb[lane * 512 + j]);
        float s = 0.f;
        for (int j = 0; j < 512; ++j) s += __expf(Sb[lane * 512 + j] - m);
        rmax[lane] = m;
        rinv[lane] = 1.f / s;
    }
    __syncthreads();
    // P = softmax * m2; spill direction-masked P^T (bf16) for the F2 pass
    size_t h0 = (size_t)(b * 8 + c);
    size_t h1 = (size_t)(64 + b * 8 + c);
    for (int idx = lane; idx < 16 * 512; idx += 32) {
        int il = idx >> 9, j = idx & 511;
        int i = it * 16 + il;
        float p = __expf(Sb[idx] - rmax[il]) * rinv[il];
        if (mask[b * 512 + i] == 0 || mask[b * 512 + j] == 0) p = 0.f;
        Sb[idx] = p;
        PTm[(h0 * 512 + j) * 512 + i] = f2bf((j > i) ? p : 0.f);
        PTm[(h1 * 512 + j) * 512 + i] = f2bf((j < i) ? p : 0.f);
    }
    __syncthreads();

    // F1[i,a] = sum_{k,j} (P*dm_k)[i,j] * t2[k,b,c,j,a]   via P @ t2T
    v8f acc0[4], acc1[4];
#pragma unroll
    for (int at = 0; at < 4; ++at) { acc0[at] = zero8(); acc1[at] = zero8(); }
    int m = lane & 15;
    int iG = it * 16 + m;
    for (int kc = 0; kc < 16; ++kc) {
        v16bf f0, f1;
#pragma unroll
        for (int t = 0; t < 8; ++t) {
            int j0 = kc * 32 + hb + t;
            float p = Sb[m * 512 + j0];
            f0[t] = f2bf((j0 > iG) ? p : 0.f);
            f1[t] = f2bf((j0 < iG) ? p : 0.f);
            int j1 = j0 + 16;
            float q = Sb[m * 512 + j1];
            f0[t + 8] = f2bf((j1 > iG) ? q : 0.f);
            f1[t + 8] = f2bf((j1 < iG) ? q : 0.f);
        }
#pragma unroll
        for (int at = 0; at < 4; ++at) {
            const __bf16* Bt0 = t2T + ((size_t)(0 * 8 + b) * 512 + c * 64 + at * 16) * 512;
            const __bf16* Bt1 = t2T + ((size_t)(1 * 8 + b) * 512 + c * 64 + at * 16) * 512;
            acc0[at] = wmma_bf(f0, load_bT(Bt0, 512, kc * 32), acc0[at]);
            acc1[at] = wmma_bf(f1, load_bT(Bt1, 512, kc * 32), acc1[at]);
        }
    }
#pragma unroll
    for (int at = 0; at < 4; ++at)
#pragma unroll
        for (int r = 0; r < 8; ++r) {
            int i = it * 16 + r + hb, a = at * 16 + n;
            atomicAdd(&fz[((size_t)b * 512 + i) * 64 + a], acc0[at][r] + acc1[at][r]);
        }
}

// ------- F2[j,e] = sum_{k,c,i} (P*dm_k)^T[j,i] * t1[k,b,c,i,e] ------------------
__global__ void k_f2(const __bf16* __restrict__ PTm, const __bf16* __restrict__ t1T,
                     float* fz) {
    int jt = blockIdx.x, c = blockIdx.y, b = blockIdx.z, lane = threadIdx.x;
    int n = lane & 15, hb = (lane >> 4) << 3;
    const __bf16* A0 = PTm + ((size_t)(b * 8 + c) * 512 + jt * 16) * 512;
    const __bf16* A1 = PTm + ((size_t)(64 + b * 8 + c) * 512 + jt * 16) * 512;
    v8f acc[4];
#pragma unroll
    for (int et = 0; et < 4; ++et) acc[et] = zero8();
    for (int ic = 0; ic < 16; ++ic) {
        if (ic + 1 < 16) {
            __builtin_prefetch(A0 + (lane & 15) * 512 + (ic + 1) * 32, 0, 1);
            __builtin_prefetch(A1 + (lane & 15) * 512 + (ic + 1) * 32, 0, 1);
        }
        v16bf f0 = load_aRM(A0, 512, ic * 32);
        v16bf f1 = load_aRM(A1, 512, ic * 32);
#pragma unroll
        for (int et = 0; et < 4; ++et) {
            const __bf16* B0 = t1T + ((size_t)(0 * 8 + b) * 512 + c * 64 + et * 16) * 512;
            const __bf16* B1 = t1T + ((size_t)(1 * 8 + b) * 512 + c * 64 + et * 16) * 512;
            acc[et] = wmma_bf(f0, load_bT(B0, 512, ic * 32), acc[et]);
            acc[et] = wmma_bf(f1, load_bT(B1, 512, ic * 32), acc[et]);
        }
    }
#pragma unroll
    for (int et = 0; et < 4; ++et)
#pragma unroll
        for (int r = 0; r < 8; ++r) {
            int j = jt * 16 + r + hb, e = et * 16 + n;
            atomicAdd(&fz[((size_t)b * 512 + j) * 64 + e], acc[et][r]);
        }
}

// ------- global head: Fg = softmax(q_z @ gw) @ gw^T -----------------------------
__global__ void k_glob(const __bf16* __restrict__ qzb, const __bf16* __restrict__ gwb,
                       const __bf16* __restrict__ gwT, const int* __restrict__ mask,
                       float* fz) {
    __shared__ float Gs[16 * 64];
    __shared__ __bf16 Gb[16 * 64];
    int it = blockIdx.x, b = blockIdx.y, lane = threadIdx.x;
    int n = lane & 15, hb = (lane >> 4) << 3;
    const __bf16* A = qzb + ((size_t)b * 512 + it * 16) * 64;
    v16bf a0 = load_aRM(A, 64, 0), a1 = load_aRM(A, 64, 32);
#pragma unroll
    for (int gt = 0; gt < 4; ++gt) {
        v8f g = zero8();
        g = wmma_bf(a0, load_bT(gwb + gt * 16 * 64, 64, 0), g);
        g = wmma_bf(a1, load_bT(gwb + gt * 16 * 64, 64, 32), g);
#pragma unroll
        for (int r = 0; r < 8; ++r) Gs[(r + hb) * 64 + gt * 16 + n] = g[r];
    }
    __syncthreads();
    if (lane < 16) {
        int i = it * 16 + lane;
        float m = -1e30f;
        for (int g = 0; g < 64; ++g) m = fmaxf(m, Gs[lane * 64 + g]);
        float s = 0.f;
        for (int g = 0; g < 64; ++g) s += __expf(Gs[lane * 64 + g] - m);
        float inv = ((mask[b * 512 + i] != 0) ? 1.f : 0.f) / s;
        for (int g = 0; g < 64; ++g)
            Gb[lane * 64 + g] = f2bf(__expf(Gs[lane * 64 + g] - m) * inv);
    }
    __syncthreads();
    v16bf q0 = load_aRM((const __bf16*)Gb, 64, 0);
    v16bf q1 = load_aRM((const __bf16*)Gb, 64, 32);
#pragma unroll
    for (int at = 0; at < 4; ++at) {
        v8f f = zero8();
        f = wmma_bf(q0, load_bT(gwT + at * 16 * 64, 64, 0), f);
        f = wmma_bf(q1, load_bT(gwT + at * 16 * 64, 64, 32), f);
#pragma unroll
        for (int r = 0; r < 8; ++r) {
            int i = it * 16 + r + hb, a = at * 16 + n;
            atomicAdd(&fz[((size_t)b * 512 + i) * 64 + a], f[r]);
        }
    }
}

extern "C" void kernel_launch(void* const* d_in, const int* in_sizes, int n_in,
                              void* d_out, int out_size, void* d_ws, size_t ws_size,
                              hipStream_t stream) {
    const float* x    = (const float*)d_in[0];   // [8,512,64]
    const int*   mask = (const int*)d_in[1];     // [8,512]
    const float* tern = (const float*)d_in[2];   // [2,64,64,8]
    const float* gw   = (const float*)d_in[3];   // [64,64,1]

    char* ws = (char*)d_ws;
    size_t o = 0;
    auto alloc = [&](size_t bytes) -> void* {
        void* p = ws + o;
        o += (bytes + 255) & ~(size_t)255;
        return p;
    };
    __bf16* qzb = (__bf16*)alloc((size_t)NB * NL * ND * 2);
    __bf16* t1  = (__bf16*)alloc((size_t)2 * NB * NL * 512 * 2);
    __bf16* t1T = (__bf16*)alloc((size_t)2 * NB * NL * 512 * 2);
    __bf16* t2T = (__bf16*)alloc((size_t)2 * NB * NL * 512 * 2);
    __bf16* PTm = (__bf16*)alloc((size_t)2 * NB * NC * NL * NL * 2);
    __bf16* W1T = (__bf16*)alloc((size_t)2 * 512 * 64 * 2);
    __bf16* W2T = (__bf16*)alloc((size_t)2 * 512 * 64 * 2);
    __bf16* gwb = (__bf16*)alloc((size_t)64 * 64 * 2);
    __bf16* gwT = (__bf16*)alloc((size_t)64 * 64 * 2);
    float*  fz0 = (float*)alloc((size_t)NB * NL * ND * 4);
    float*  fz1 = (float*)alloc((size_t)NB * NL * ND * 4);

    k_prep<<<256, 256, 0, stream>>>(tern, gw, W1T, W2T, gwb, gwT);

    const float* zin = x;
    for (int iter = 0; iter < 3; ++iter) {
        float* out = (iter == 0) ? fz0 : (iter == 1) ? fz1 : (float*)d_out;
        k_qz<<<NB * NL, 32, 0, stream>>>(zin, mask, qzb);
        k_t1t2<<<dim3(32, 32, 16), 32, 0, stream>>>(qzb, W1T, W2T, t1, t1T, t2T);
        k_fzinit<<<1024, 256, 0, stream>>>(out, x);
        k_attn<<<dim3(32, NC, NB), 32, 0, stream>>>(t1, qzb, t2T, mask, PTm, out);
        k_f2<<<dim3(32, NC, NB), 32, 0, stream>>>(PTm, t1T, out);
        k_glob<<<dim3(32, NB), 32, 0, stream>>>(qzb, gwb, gwT, mask, out);
        zin = out;
    }
}
